// GraphNetSurfaceBlockMHA_Dense_85487029060204
// MI455X (gfx1250) — matrix operand order repaired and checked by reading, not compiled, not run on hardware
//
#include <hip/hip_runtime.h>
#include <hip/hip_bf16.h>
#include <math.h>

#define NODES 4096
#define HID   256
#define NHEADS 8
#define HDIM  32
#define MASKW (NODES / 32) /* 128 mask words per row */

typedef __attribute__((ext_vector_type(16))) __bf16 bf16x16;
typedef __attribute__((ext_vector_type(8)))  float  f32x8;

union Frag16 {
    bf16x16        bf;
    unsigned int   u[8];
    unsigned short s[16];
};

__device__ __forceinline__ unsigned short f2bf(float f) {
    unsigned int u = __float_as_uint(f);
    u += 0x7FFFu + ((u >> 16) & 1u);      // round-to-nearest-even
    return (unsigned short)(u >> 16);
}

__device__ __forceinline__ float gelu_exact(float x) {
    return 0.5f * x * (1.0f + erff(x * 0.70710678118654752f));
}

// ---------------------------------------------------------------------------
// node_in MLP: x = GELU(pos @ W1^T + b1) @ W2^T + b2, emitted as bf16
// ---------------------------------------------------------------------------
__global__ void node_mlp_kernel(const float* __restrict__ pos,
                                const float* __restrict__ W1, const float* __restrict__ b1,
                                const float* __restrict__ W2, const float* __restrict__ b2,
                                unsigned short* __restrict__ xbf) {
    __shared__ float p3[3];
    __shared__ float h1[HID];
    const int n = blockIdx.x, t = threadIdx.x;
    if (t < 3) p3[t] = pos[n * 3 + t];
    __syncthreads();
    float a = b1[t] + p3[0] * W1[t * 3 + 0] + p3[1] * W1[t * 3 + 1] + p3[2] * W1[t * 3 + 2];
    h1[t] = gelu_exact(a);
    __syncthreads();
    float s = b2[t];
    const float* wr = W2 + (size_t)t * HID;
    for (int k = 0; k < HID; ++k) s += h1[k] * wr[k];
    xbf[(size_t)n * HID + t] = f2bf(s);
}

// ---------------------------------------------------------------------------
// helpers: f32 -> bf16 conversion, mask init, mask scatter
// ---------------------------------------------------------------------------
__global__ void f2bf_kernel(const float* __restrict__ src, unsigned short* __restrict__ dst, int n) {
    int i = blockIdx.x * blockDim.x + threadIdx.x;
    if (i < n) dst[i] = f2bf(src[i]);
}

__global__ void zero_u32_kernel(unsigned* __restrict__ p, int n) {
    int i = blockIdx.x * blockDim.x + threadIdx.x;
    if (i < n) p[i] = 0u;
}

__global__ void build_mask_kernel(const int* __restrict__ ei, int E, unsigned* __restrict__ mb) {
    int e = blockIdx.x * blockDim.x + threadIdx.x;
    if (e < E) {
        unsigned i = ((unsigned)ei[e]) & (NODES - 1);
        unsigned j = ((unsigned)ei[E + e]) & (NODES - 1);
        atomicOr(&mb[(size_t)i * MASKW + (j >> 5)], 1u << (j & 31u));
    } else if (e < E + NODES) {
        unsigned i = (unsigned)(e - E);
        atomicOr(&mb[(size_t)i * MASKW + (i >> 5)], 1u << (i & 31u));
    }
}

// ---------------------------------------------------------------------------
// QKV projection: qkv = x @ in_proj_w^T + b  (bf16 WMMA, f32 accumulate)
// Writes q,k row-major per head [h][node][d]; v transposed per head [h][d][node]
// ---------------------------------------------------------------------------
__global__ void qkv_wmma_kernel(const unsigned short* __restrict__ xbf,
                                const unsigned short* __restrict__ wbf,
                                const float* __restrict__ bias,
                                unsigned short* __restrict__ q,
                                unsigned short* __restrict__ k,
                                unsigned short* __restrict__ vt) {
    const int lane = threadIdx.x;
    const int m = lane & 15, hi = lane >> 4;
    const int ib = blockIdx.x;   // node tile (N/16)
    const int jb = blockIdx.y;   // output-channel tile (768/16)

    f32x8 acc = {0.f, 0.f, 0.f, 0.f, 0.f, 0.f, 0.f, 0.f};
    const unsigned short* xrow = xbf + (size_t)(ib * 16 + m) * HID;
    const unsigned short* wrow = wbf + (size_t)(jb * 16 + m) * HID;

    for (int kb = 0; kb < HID; kb += 32) {
        Frag16 a, b;
#pragma unroll
        for (int r = 0; r < 8; ++r) {
            const int kk = hi * 16 + r * 2; // A contiguous
            a.u[r] = *(const unsigned int*)(xrow + kb + kk);
            b.u[r] = *(const unsigned int*)(wrow + kb + hi * 16 + r * 2); // B contiguous
        }
        acc = __builtin_amdgcn_wmma_f32_16x16x32_bf16(false, a.bf, false, b.bf,
                                                      (short)0, acc, false, false);
    }

    const int c = jb * 16 + m;
    const float bv = bias[c];
#pragma unroll
    for (int r = 0; r < 8; ++r) {
        const int row = ib * 16 + r + 8 * hi;
        const unsigned short v = f2bf(acc[r] + bv);
        if (c < 256) {
            const int h = c >> 5, d = c & 31;
            q[((size_t)h * NODES + row) * HDIM + d] = v;
        } else if (c < 512) {
            const int c2 = c - 256, h = c2 >> 5, d = c2 & 31;
            k[((size_t)h * NODES + row) * HDIM + d] = v;
        } else {
            const int c2 = c - 512, h = c2 >> 5, d = c2 & 31;
            vt[((size_t)h * HDIM + d) * NODES + row] = v;
        }
    }
}

// ---------------------------------------------------------------------------
// Masked flash attention: 1 wave per (head, 32-query tile); 32 keys / step.
// 8 WMMAs per key tile reusing one set of K/V B-fragments (2x intensity).
// ---------------------------------------------------------------------------
__global__ void flash_attn_kernel(const unsigned short* __restrict__ q,
                                  const unsigned short* __restrict__ kmat,
                                  const unsigned short* __restrict__ vt,
                                  const unsigned* __restrict__ maskw,
                                  float* __restrict__ ctx) {
    __shared__ unsigned short p_lds[32][32];
    const int lane = threadIdx.x;
    const int m = lane & 15, hi = lane >> 4;
    const int qb = blockIdx.x * 32;
    const int h  = blockIdx.y;
    const float scale = 0.17677669529663687f; // 1/sqrt(32)
    const float NEG = -__builtin_huge_valf();

    // two invariant Q fragments (A layout): rows qb..qb+15 and qb+16..qb+31
    Frag16 aq[2];
#pragma unroll
    for (int p = 0; p < 2; ++p) {
        const unsigned short* qrow = q + ((size_t)h * NODES + qb + p * 16 + m) * HDIM;
#pragma unroll
        for (int r = 0; r < 8; ++r) {
            const int kk = hi * 16 + r * 2;
            aq[p].u[r] = *(const unsigned int*)(qrow + kk);
        }
    }

    float mrow[2][8], lrow[2][8];
    f32x8 acc0[2], acc1[2];
#pragma unroll
    for (int p = 0; p < 2; ++p) {
#pragma unroll
        for (int r = 0; r < 8; ++r) {
            mrow[p][r] = NEG; lrow[p][r] = 0.f;
            acc0[p][r] = 0.f; acc1[p][r] = 0.f;
        }
    }

    const unsigned short* kh = kmat + (size_t)h * NODES * HDIM;
    const unsigned short* vh = vt   + (size_t)h * HDIM * NODES;

    for (int kt = 0; kt < NODES / 32; ++kt) {
        const int kb = kt * 32;

        // prefetch next key tile's K rows + V columns (global_prefetch_b8)
        if (kt + 1 < NODES / 32) {
            __builtin_prefetch(kh + (size_t)(kb + 32) * HDIM + lane * 32, 0, 1);
            __builtin_prefetch(vh + (size_t)lane * NODES + kb + 32, 0, 1);
        }

        unsigned mw[2][8]; unsigned any = 0;
#pragma unroll
        for (int p = 0; p < 2; ++p)
#pragma unroll
            for (int r = 0; r < 8; ++r) {
                mw[p][r] = maskw[(size_t)(qb + p * 16 + r + 8 * hi) * MASKW + kt];
                any |= mw[p][r];
            }
        if (!__any(any != 0u)) continue;   // fully-masked key tile

        Frag16 bk0, bk1;
#pragma unroll
        for (int r = 0; r < 8; ++r) {
            bk0.u[r] = *(const unsigned int*)(kh + (size_t)(kb +      m) * HDIM + hi * 16 + r * 2);
            bk1.u[r] = *(const unsigned int*)(kh + (size_t)(kb + 16 + m) * HDIM + hi * 16 + r * 2);
        }
        const f32x8 z = {0.f, 0.f, 0.f, 0.f, 0.f, 0.f, 0.f, 0.f};
        f32x8 s0[2], s1[2];
#pragma unroll
        for (int p = 0; p < 2; ++p) {
            s0[p] = __builtin_amdgcn_wmma_f32_16x16x32_bf16(false, aq[p].bf, false, bk0.bf,
                                                            (short)0, z, false, false);
            s1[p] = __builtin_amdgcn_wmma_f32_16x16x32_bf16(false, aq[p].bf, false, bk1.bf,
                                                            (short)0, z, false, false);
        }

        // online softmax for 32 rows (two row-groups of 16)
#pragma unroll
        for (int p = 0; p < 2; ++p) {
#pragma unroll
            for (int r = 0; r < 8; ++r) {
                float x0 = ((mw[p][r] >> m) & 1u)        ? s0[p][r] * scale : NEG;
                float x1 = ((mw[p][r] >> (16 + m)) & 1u) ? s1[p][r] * scale : NEG;
                float t = fmaxf(x0, x1);
#pragma unroll
                for (int off = 1; off < 16; off <<= 1) t = fmaxf(t, __shfl_xor(t, off, 16));
                const float mnew = fmaxf(mrow[p][r], t);
                float corr, p0, p1;
                if (mnew == NEG) { corr = 1.0f; p0 = 0.0f; p1 = 0.0f; }
                else {
                    corr = (mrow[p][r] == NEG) ? 0.0f : __expf(mrow[p][r] - mnew);
                    p0 = (x0 == NEG) ? 0.0f : __expf(x0 - mnew);
                    p1 = (x1 == NEG) ? 0.0f : __expf(x1 - mnew);
                }
                float ps = p0 + p1;
#pragma unroll
                for (int off = 1; off < 16; off <<= 1) ps += __shfl_xor(ps, off, 16);
                lrow[p][r] = lrow[p][r] * corr + ps;
                mrow[p][r] = mnew;
                acc0[p][r] *= corr;
                acc1[p][r] *= corr;
                p_lds[p * 16 + r + 8 * hi][m]      = f2bf(p0);
                p_lds[p * 16 + r + 8 * hi][16 + m] = f2bf(p1);
            }
        }
        __syncthreads();   // single-wave block: cheap; orders LDS P-transpose

        Frag16 bv0, bv1;
#pragma unroll
        for (int r = 0; r < 8; ++r) {
            bv0.u[r] = *(const unsigned int*)(vh + (size_t)m        * NODES + kb + hi * 16 + r * 2);
            bv1.u[r] = *(const unsigned int*)(vh + (size_t)(16 + m) * NODES + kb + hi * 16 + r * 2);
        }
#pragma unroll
        for (int p = 0; p < 2; ++p) {
            Frag16 ap;
#pragma unroll
            for (int r = 0; r < 8; ++r) {
                const int kk = hi * 16 + r * 2;
                ap.u[r] = *(const unsigned int*)&p_lds[p * 16 + m][kk];
            }
            acc0[p] = __builtin_amdgcn_wmma_f32_16x16x32_bf16(false, ap.bf, false, bv0.bf,
                                                              (short)0, acc0[p], false, false);
            acc1[p] = __builtin_amdgcn_wmma_f32_16x16x32_bf16(false, ap.bf, false, bv1.bf,
                                                              (short)0, acc1[p], false, false);
        }
        __syncthreads();
    }

#pragma unroll
    for (int p = 0; p < 2; ++p)
#pragma unroll
        for (int r = 0; r < 8; ++r) {
            const float inv = (lrow[p][r] > 0.0f) ? 1.0f / lrow[p][r] : 0.0f;
            const int row = qb + p * 16 + r + 8 * hi;
            ctx[(size_t)row * HID + h * HDIM + m]      = acc0[p][r] * inv;
            ctx[(size_t)row * HID + h * HDIM + 16 + m] = acc1[p][r] * inv;
        }
}

// ---------------------------------------------------------------------------
// out_proj + out_net fused (f32): y = GELU([ctx@Wo^T+bo, pos] @ W1^T + b1) @ W2^T + b2
// ---------------------------------------------------------------------------
__global__ void out_net_kernel(const float* __restrict__ ctx,
                               const float* __restrict__ pos,
                               const float* __restrict__ Wo, const float* __restrict__ bo,
                               const float* __restrict__ W1, const float* __restrict__ b1,
                               const float* __restrict__ W2, const float* __restrict__ b2,
                               float* __restrict__ out) {
    __shared__ float tmp[HID + 3];
    __shared__ float hh[HID];
    const int n = blockIdx.x, t = threadIdx.x;
    const float* crow = ctx + (size_t)n * HID;
    float o = bo[t];
    const float* wr = Wo + (size_t)t * HID;
    for (int k = 0; k < HID; ++k) o += crow[k] * wr[k];
    tmp[t] = o;
    if (t < 3) tmp[HID + t] = pos[n * 3 + t];
    __syncthreads();
    float a = b1[t];
    const float* w1r = W1 + (size_t)t * (HID + 3);
    for (int k = 0; k < HID + 3; ++k) a += tmp[k] * w1r[k];
    hh[t] = gelu_exact(a);
    __syncthreads();
    if (t < 3) {
        float y = b2[t];
        const float* w2r = W2 + (size_t)t * HID;
        for (int k = 0; k < HID; ++k) y += hh[k] * w2r[k];
        out[(size_t)n * 3 + t] = y;
    }
}

// ---------------------------------------------------------------------------
extern "C" void kernel_launch(void* const* d_in, const int* in_sizes, int n_in,
                              void* d_out, int out_size, void* d_ws, size_t ws_size,
                              hipStream_t stream) {
    const float* pos        = (const float*)d_in[0];
    const int*   ei         = (const int*)d_in[1];
    const float* W_in1      = (const float*)d_in[2];
    const float* b_in1      = (const float*)d_in[3];
    const float* W_in2      = (const float*)d_in[4];
    const float* b_in2      = (const float*)d_in[5];
    const float* in_proj_w  = (const float*)d_in[6];
    const float* in_proj_b  = (const float*)d_in[7];
    const float* out_proj_w = (const float*)d_in[8];
    const float* out_proj_b = (const float*)d_in[9];
    const float* W_o1       = (const float*)d_in[10];
    const float* b_o1       = (const float*)d_in[11];
    const float* W_o2       = (const float*)d_in[12];
    const float* b_o2       = (const float*)d_in[13];
    float* out = (float*)d_out;
    const int E = in_sizes[1] / 2;

    // carve workspace
    char* ws = (char*)d_ws;
    size_t off = 0;
    auto carve = [&](size_t bytes) -> char* {
        char* p = ws + off;
        off += (bytes + 255) & ~(size_t)255;
        return p;
    };
    unsigned short* xbf   = (unsigned short*)carve((size_t)NODES * HID * 2);
    unsigned short* wbf   = (unsigned short*)carve((size_t)3 * HID * HID * 2);
    unsigned short* qbuf  = (unsigned short*)carve((size_t)NHEADS * NODES * HDIM * 2);
    unsigned short* kbuf  = (unsigned short*)carve((size_t)NHEADS * NODES * HDIM * 2);
    unsigned short* vtbuf = (unsigned short*)carve((size_t)NHEADS * HDIM * NODES * 2);
    unsigned*       maskb = (unsigned*)carve((size_t)NODES * MASKW * 4);
    float*          ctx   = (float*)carve((size_t)NODES * HID * 4);

    node_mlp_kernel<<<NODES, HID, 0, stream>>>(pos, W_in1, b_in1, W_in2, b_in2, xbf);

    const int nw = 3 * HID * HID;
    f2bf_kernel<<<(nw + 255) / 256, 256, 0, stream>>>(in_proj_w, wbf, nw);

    const int nm = NODES * MASKW;
    zero_u32_kernel<<<(nm + 255) / 256, 256, 0, stream>>>(maskb, nm);
    build_mask_kernel<<<(E + NODES + 255) / 256, 256, 0, stream>>>(ei, E, maskb);

    qkv_wmma_kernel<<<dim3(NODES / 16, 3 * HID / 16), 32, 0, stream>>>(
        xbf, wbf, in_proj_b, qbuf, kbuf, vtbuf);

    flash_attn_kernel<<<dim3(NODES / 32, NHEADS), 32, 0, stream>>>(
        qbuf, kbuf, vtbuf, maskb, ctx);

    out_net_kernel<<<NODES, HID, 0, stream>>>(
        ctx, pos, out_proj_w, out_proj_b, W_o1, b_o1, W_o2, b_o2, out);
}
